// SNN_VGG9_BNTT_DCLS_41781441855721
// MI455X (gfx1250) — matrix operand order repaired and checked by reading, not compiled, not run on hardware
//
#include <hip/hip_runtime.h>
#include <cstdint>
#include <cstddef>

typedef __attribute__((ext_vector_type(16))) _Float16 v16h;
typedef __attribute__((ext_vector_type(8)))  _Float16 v8h;
typedef __attribute__((ext_vector_type(8)))  float    v8f;
typedef int v4i __attribute__((vector_size(16)));   // matches builtin signature

#define LEAK  0.95f
#define BNEPS 1e-4f

// ---- CDNA5 async global->LDS copy (gfx1250), with safe fallback ----------
#ifndef __has_builtin
#define __has_builtin(x) 0
#endif
#if __has_builtin(__builtin_amdgcn_global_load_async_to_lds_b128)
#define HAVE_ASYNC 1
#define ASYNC_CP16(g, l)                                                     \
  __builtin_amdgcn_global_load_async_to_lds_b128(                            \
      (__attribute__((address_space(1))) v4i*)(g),                           \
      (__attribute__((address_space(3))) v4i*)(l), 0, 0)
#else
#define HAVE_ASYNC 0
#endif

#if __has_builtin(__builtin_amdgcn_s_wait_asynccnt)
#define WAIT_ASYNC0() __builtin_amdgcn_s_wait_asynccnt(0)
#else
#define WAIT_ASYNC0() asm volatile("s_wait_asynccnt 0x0" ::: "memory")
#endif

// ---------------------------------------------------------------------------
// Utility kernels
// ---------------------------------------------------------------------------

__global__ void k_zero(float* __restrict__ p, size_t n) {
  size_t i = (size_t)blockIdx.x * blockDim.x + threadIdx.x;
  size_t stride = (size_t)gridDim.x * blockDim.x;
  for (; i < n; i += stride) p[i] = 0.f;
}

// Convert f32 weight matrix [M][K] row-major -> f16 [M][Kpad], zero-padded.
__global__ void k_cvt_w(const float* __restrict__ w, _Float16* __restrict__ o,
                        int M, int K, int Kpad) {
  int i = blockIdx.x * blockDim.x + threadIdx.x;
  if (i >= M * Kpad) return;
  int k = i % Kpad, m = i / Kpad;
  o[i] = (k < K) ? (_Float16)w[(size_t)m * K + k] : (_Float16)0.f;
}

// Stateless hash -> uniform [0,1)
__device__ __forceinline__ float u01hash(uint32_t v) {
  v ^= v >> 16; v *= 0x7feb352dU;
  v ^= v >> 15; v *= 0x846ca68bU;
  v ^= v >> 16;
  return (float)(v >> 8) * (1.0f / 16777216.0f);
}

// Poisson-ish spike encoding of the input image.
__global__ void k_spike_in(const float* __restrict__ inp, float* __restrict__ sp,
                           int total, int t) {
  int i = blockIdx.x * blockDim.x + threadIdx.x;
  if (i >= total) return;
  float x = inp[i];
  float r = u01hash((uint32_t)i * 2654435761u ^ (0x9e3779b9u * (uint32_t)(t + 1)));
  float sgn = (x > 0.f) ? 1.f : ((x < 0.f) ? -1.f : 0.f);
  sp[i] = (r * 0.5f <= fabsf(x)) ? sgn : 0.f;
}

// im2col (3x3, pad 1): x f32 NCHW -> P f16 [N=B*H*W][Kpad], k = c*9 + ky*3+kx
__global__ void k_im2col(const float* __restrict__ x, _Float16* __restrict__ P,
                         int B, int C, int H, int W, int Kpad) {
  int idx = blockIdx.x * blockDim.x + threadIdx.x;
  int HW = H * W;
  int total = B * HW * C;
  if (idx >= total) return;
  int c = idx % C;
  int n = idx / C;          // n = b*HW + hw
  int hw = n % HW, b = n / HW;
  int h = hw / W, w = hw % W;
  const float* xc = x + ((size_t)b * C + c) * HW;
  _Float16* row = P + (size_t)n * Kpad + c * 9;
#pragma unroll
  for (int ky = 0; ky < 3; ++ky)
#pragma unroll
    for (int kx = 0; kx < 3; ++kx) {
      int yy = h + ky - 1, xx = w + kx - 1;
      float v = (yy >= 0 && yy < H && xx >= 0 && xx < W) ? xc[yy * W + xx] : 0.f;
      row[ky * 3 + kx] = (_Float16)v;
    }
}

// Deformable bilinear gather: x f32 NCHW, off f32 [B][18][HW]
// -> P f16 [N=B*HW][Kpad], k = c*9 + k2
__global__ void k_deform(const float* __restrict__ x, const float* __restrict__ off,
                         _Float16* __restrict__ P, int B, int C, int H, int W,
                         int Kpad) {
  int idx = blockIdx.x * blockDim.x + threadIdx.x;
  int HW = H * W;
  int total = B * 9 * HW;
  if (idx >= total) return;
  int hw = idx % HW;
  int tt = idx / HW;
  int k2 = tt % 9, b = tt / 9;
  int h = hw / W, w = hw % W;
  int ky = k2 / 3, kx = k2 % 3;
  float oy = off[(((size_t)b * 18 + k2 * 2 + 0) * HW) + hw];
  float ox = off[(((size_t)b * 18 + k2 * 2 + 1) * HW) + hw];
  float py = (float)h + (float)(ky - 1) + oy;
  float px = (float)w + (float)(kx - 1) + ox;
  float y0f = floorf(py), x0f = floorf(px);
  float wy = py - y0f, wx = px - x0f;
  int y0 = (int)y0f, x0 = (int)x0f;
  int y1 = y0 + 1, x1 = x0 + 1;
  bool vy0 = (y0 >= 0) && (y0 < H), vy1 = (y1 >= 0) && (y1 < H);
  bool vx0 = (x0 >= 0) && (x0 < W), vx1 = (x1 >= 0) && (x1 < W);
  int cy0 = min(max(y0, 0), H - 1), cy1 = min(max(y1, 0), H - 1);
  int cx0 = min(max(x0, 0), W - 1), cx1 = min(max(x1, 0), W - 1);
  float w00 = (1.f - wy) * (1.f - wx) * ((vy0 && vx0) ? 1.f : 0.f);
  float w01 = (1.f - wy) * wx         * ((vy0 && vx1) ? 1.f : 0.f);
  float w10 = wy * (1.f - wx)         * ((vy1 && vx0) ? 1.f : 0.f);
  float w11 = wy * wx                 * ((vy1 && vx1) ? 1.f : 0.f);
  int i00 = cy0 * W + cx0, i01 = cy0 * W + cx1;
  int i10 = cy1 * W + cx0, i11 = cy1 * W + cx1;
  _Float16* row = P + (size_t)(b * HW + hw) * Kpad + k2;
  const float* xb = x + (size_t)b * C * HW;
  for (int c = 0; c < C; ++c) {
    const float* xc = xb + (size_t)c * HW;
    float v = w00 * xc[i00] + w01 * xc[i01] + w10 * xc[i10] + w11 * xc[i11];
    row[(size_t)c * 9] = (_Float16)v;
  }
}

// Per-channel mean/var (biased) over (B,HW). One block per channel.
__global__ void k_bnstats(const float* __restrict__ y, float* __restrict__ mean,
                          float* __restrict__ var, int C, int NHW, int HW) {
  int c = blockIdx.x;
  float s = 0.f, s2 = 0.f;
  for (int i = threadIdx.x; i < NHW; i += blockDim.x) {
    int b = i / HW, hw = i % HW;
    float v = y[((size_t)b * C + c) * HW + hw];
    s += v; s2 += v * v;
  }
  __shared__ float sh[256], sh2[256];
  sh[threadIdx.x] = s; sh2[threadIdx.x] = s2;
  __syncthreads();
  for (int st = 128; st > 0; st >>= 1) {
    if ((int)threadIdx.x < st) {
      sh[threadIdx.x] += sh[threadIdx.x + st];
      sh2[threadIdx.x] += sh2[threadIdx.x + st];
    }
    __syncthreads();
  }
  if (threadIdx.x == 0) {
    float m = sh[0] / (float)NHW;
    mean[c] = m;
    var[c] = sh2[0] / (float)NHW - m * m;
  }
}

// Fused BN-apply + leaky membrane + spike (2d feature maps).
__global__ void k_mem2d(const float* __restrict__ y, const float* __restrict__ mean,
                        const float* __restrict__ var, const float* __restrict__ g,
                        const float* __restrict__ bt, float* __restrict__ m,
                        float* __restrict__ s, int C, int HW, int total) {
  int i = blockIdx.x * blockDim.x + threadIdx.x;
  if (i >= total) return;
  int c = (i / HW) % C;
  float norm = (y[i] - mean[c]) * (g[c] * rsqrtf(var[c] + BNEPS)) + bt[c];
  float mm = LEAK * m[i] + norm;
  m[i] = mm;
  s[i] = (mm - 1.0f > 0.f) ? 1.f : 0.f;
}

// 2x2 average pool, f32 -> f32 (contiguous over (b,c)).
__global__ void k_pool(const float* __restrict__ s, float* __restrict__ o,
                       int H, int W, int total) {
  int i = blockIdx.x * blockDim.x + threadIdx.x;
  if (i >= total) return;
  int OH = H / 2, OW = W / 2;
  int ohw = i % (OH * OW);
  int bc = i / (OH * OW);
  int oh = ohw / OW, ow = ohw % OW;
  const float* src = s + ((size_t)bc * H + 2 * oh) * W + 2 * ow;
  o[i] = 0.25f * (src[0] + src[1] + src[W] + src[W + 1]);
}

// 2x2 pool + flatten to f16 fc patch matrix [b][k = c*OH*OW + ohw]
__global__ void k_pool_fc(const float* __restrict__ s, _Float16* __restrict__ P,
                          int C, int H, int W, int Kpad, int total) {
  int i = blockIdx.x * blockDim.x + threadIdx.x;
  if (i >= total) return;
  int OH = H / 2, OW = W / 2;
  int ohw = i % (OH * OW);
  int c = (i / (OH * OW)) % C;
  int b = i / (OH * OW * C);
  int oh = ohw / OW, ow = ohw % OW;
  const float* src = s + (((size_t)b * C + c) * H + 2 * oh) * W + 2 * ow;
  float v = 0.25f * (src[0] + src[1] + src[W] + src[W + 1]);
  P[(size_t)b * Kpad + c * (OH * OW) + ohw] = (_Float16)v;
}

// BN1d stats over batch dim: z [B][F]
__global__ void k_bnstats1d(const float* __restrict__ z, float* __restrict__ mean,
                            float* __restrict__ var, int F, int B) {
  int f = blockIdx.x * blockDim.x + threadIdx.x;
  if (f >= F) return;
  float s = 0.f, s2 = 0.f;
  for (int b = 0; b < B; ++b) {
    float v = z[(size_t)b * F + f];
    s += v; s2 += v * v;
  }
  float m = s / (float)B;
  mean[f] = m;
  var[f] = s2 / (float)B - m * m;
}

// Fused BN1d + membrane + spike -> f16 fc2 input [b][f]
__global__ void k_memfc(const float* __restrict__ z, const float* __restrict__ mean,
                        const float* __restrict__ var, const float* __restrict__ g,
                        const float* __restrict__ bt, float* __restrict__ mf,
                        _Float16* __restrict__ S, int F, int total) {
  int i = blockIdx.x * blockDim.x + threadIdx.x;
  if (i >= total) return;
  int f = i % F;
  float norm = (z[i] - mean[f]) * (g[f] * rsqrtf(var[f] + BNEPS)) + bt[f];
  float mm = LEAK * mf[i] + norm;
  mf[i] = mm;
  S[i] = (_Float16)((mm - 1.0f > 0.f) ? 1.f : 0.f);
}

// mf2 accumulate straight into d_out
__global__ void k_accout(const float* __restrict__ tmp, float* __restrict__ out,
                         int total) {
  int i = blockIdx.x * blockDim.x + threadIdx.x;
  if (i >= total) return;
  out[i] = LEAK * out[i] + tmp[i];
}

// ---------------------------------------------------------------------------
// WMMA GEMM:  C[M x N] = A[M x Kpad] * Bp^T   (Bp stored [N][Kpad])
// Output scattered as out[b*M*HW + m*HW + hw] with n = b*HW + hw.
// Block = 128 threads = 4 waves; block tile 64(M) x 64(N); K step 32.
// Double-buffered LDS; full tiles staged with GLOBAL_LOAD_ASYNC_TO_LDS_B128
// (ASYNCcnt), edge tiles via clamped register loads + cndmask (no exec
// branching). All fragment LDS loads are issued before the 4 back-to-back
// WMMAs so the DS pipe streams while the matrix pipe drains.
// A/B fragments use contiguous k = 16*(lane/16) + s in both; D layout per ISA (wave32).
// ---------------------------------------------------------------------------
#define GT_M 64
#define GT_N 64
#define GT_K 32
#define LDT  40   // padded LDS row stride (f16 elems) to avoid bank conflicts

__global__ __launch_bounds__(128) void k_gemm_wmma(
    const _Float16* __restrict__ A, const _Float16* __restrict__ Bp,
    const float* __restrict__ bias, float* __restrict__ C,
    int M, int N, int Kpad, int HW) {
  __shared__ __align__(16) _Float16 sA[2][GT_M * LDT];
  __shared__ __align__(16) _Float16 sB[2][GT_N * LDT];

  const int tid  = threadIdx.x;
  const int lane = tid & 31;
  const int wave = tid >> 5;          // 0..3
  const int m0 = blockIdx.y * GT_M;
  const int n0 = blockIdx.x * GT_N;
  const int row = tid >> 1;           // 0..63 (staging row)
  const int ch  = tid & 1;            // staging half (16 f16 each)

  const int mrow = (wave << 4) + (lane & 15);
  const int grp  = lane >> 4;         // K-group of this lane

  v8f acc0 = {}, acc1 = {}, acc2 = {}, acc3 = {};

  const int niter = Kpad / GT_K;
  const int gm = m0 + row;
  const int gn = n0 + row;
  const int lofs = row * LDT + ch * 16;

  // Per-K-step WMMA compute: load A frag + all 4 B frags first, then issue
  // the 4 WMMAs back-to-back (single dscnt wait, XDL pipe stays busy).
  auto compute = [&](const _Float16* bufA, const _Float16* bufB) {
    v16h afrag, bf0, bf1, bf2, bf3;
    {
      const v8h lo = *(const v8h*)(bufA + mrow * LDT + grp * 16);
      const v8h hi = *(const v8h*)(bufA + mrow * LDT + grp * 16 + 8);
#pragma unroll
      for (int i = 0; i < 8; ++i) { afrag[i] = lo[i]; afrag[i + 8] = hi[i]; }
    }
    const int nb = (lane & 15);
    {
      const v8h b0 = *(const v8h*)(bufB + (0 * 16 + nb) * LDT + grp * 16);
      const v8h b1 = *(const v8h*)(bufB + (0 * 16 + nb) * LDT + grp * 16 + 8);
#pragma unroll
      for (int i = 0; i < 8; ++i) { bf0[i] = b0[i]; bf0[i + 8] = b1[i]; }
    }
    {
      const v8h b0 = *(const v8h*)(bufB + (1 * 16 + nb) * LDT + grp * 16);
      const v8h b1 = *(const v8h*)(bufB + (1 * 16 + nb) * LDT + grp * 16 + 8);
#pragma unroll
      for (int i = 0; i < 8; ++i) { bf1[i] = b0[i]; bf1[i + 8] = b1[i]; }
    }
    {
      const v8h b0 = *(const v8h*)(bufB + (2 * 16 + nb) * LDT + grp * 16);
      const v8h b1 = *(const v8h*)(bufB + (2 * 16 + nb) * LDT + grp * 16 + 8);
#pragma unroll
      for (int i = 0; i < 8; ++i) { bf2[i] = b0[i]; bf2[i + 8] = b1[i]; }
    }
    {
      const v8h b0 = *(const v8h*)(bufB + (3 * 16 + nb) * LDT + grp * 16);
      const v8h b1 = *(const v8h*)(bufB + (3 * 16 + nb) * LDT + grp * 16 + 8);
#pragma unroll
      for (int i = 0; i < 8; ++i) { bf3[i] = b0[i]; bf3[i + 8] = b1[i]; }
    }
    acc0 = __builtin_amdgcn_wmma_f32_16x16x32_f16(false, afrag, false, bf0, (short)0, acc0, false, false);
    acc1 = __builtin_amdgcn_wmma_f32_16x16x32_f16(false, afrag, false, bf1, (short)0, acc1, false, false);
    acc2 = __builtin_amdgcn_wmma_f32_16x16x32_f16(false, afrag, false, bf2, (short)0, acc2, false, false);
    acc3 = __builtin_amdgcn_wmma_f32_16x16x32_f16(false, afrag, false, bf3, (short)0, acc3, false, false);
  };

  const bool full = (m0 + GT_M <= M) && (n0 + GT_N <= N);

#if HAVE_ASYNC
  if (full) {
    const _Float16* ga = A + (size_t)gm * Kpad + ch * 16;
    const _Float16* gb = Bp + (size_t)gn * Kpad + ch * 16;
    // preload tile 0 directly into LDS (async, ASYNCcnt-tracked)
    ASYNC_CP16(ga,      &sA[0][lofs]);
    ASYNC_CP16(ga + 8,  &sA[0][lofs + 8]);
    ASYNC_CP16(gb,      &sB[0][lofs]);
    ASYNC_CP16(gb + 8,  &sB[0][lofs + 8]);
    WAIT_ASYNC0();
    __syncthreads();
    for (int it = 0; it < niter; ++it) {
      const int cur = it & 1, nxt = cur ^ 1;
      const bool more = (it + 1 < niter);
      if (more) {                         // fetch next tile while computing
        const _Float16* gan = ga + (size_t)(it + 1) * GT_K;
        const _Float16* gbn = gb + (size_t)(it + 1) * GT_K;
        ASYNC_CP16(gan,     &sA[nxt][lofs]);
        ASYNC_CP16(gan + 8, &sA[nxt][lofs + 8]);
        ASYNC_CP16(gbn,     &sB[nxt][lofs]);
        ASYNC_CP16(gbn + 8, &sB[nxt][lofs + 8]);
      }
      compute(&sA[cur][0], &sB[cur][0]);
      if (more) WAIT_ASYNC0();
      __syncthreads();
    }
  } else
#endif
  {
    // Edge tiles (or no async builtin): clamped unconditional loads + select.
    const v8h z = {};
    const int gmc = (gm < M) ? gm : (M - 1);
    const int gnc = (gn < N) ? gn : (N - 1);
    const bool okM = (gm < M);
    const bool okN = (gn < N);
    const _Float16* ga = A + (size_t)gmc * Kpad + ch * 16;
    const _Float16* gb = Bp + (size_t)gnc * Kpad + ch * 16;
    v8h ra0, ra1, rb0, rb1;
    ra0 = *(const v8h*)(ga);     ra1 = *(const v8h*)(ga + 8);
    rb0 = *(const v8h*)(gb);     rb1 = *(const v8h*)(gb + 8);
    ra0 = okM ? ra0 : z;  ra1 = okM ? ra1 : z;
    rb0 = okN ? rb0 : z;  rb1 = okN ? rb1 : z;
    *(v8h*)(&sA[0][lofs])     = ra0;
    *(v8h*)(&sA[0][lofs + 8]) = ra1;
    *(v8h*)(&sB[0][lofs])     = rb0;
    *(v8h*)(&sB[0][lofs + 8]) = rb1;
    __syncthreads();
    for (int it = 0; it < niter; ++it) {
      const int cur = it & 1, nxt = cur ^ 1;
      const bool more = (it + 1 < niter);
      if (more) {
        const _Float16* san = ga + (size_t)(it + 1) * GT_K;
        const _Float16* sbn = gb + (size_t)(it + 1) * GT_K;
        ra0 = *(const v8h*)(san);     ra1 = *(const v8h*)(san + 8);
        rb0 = *(const v8h*)(sbn);     rb1 = *(const v8h*)(sbn + 8);
        ra0 = okM ? ra0 : z;  ra1 = okM ? ra1 : z;
        rb0 = okN ? rb0 : z;  rb1 = okN ? rb1 : z;
      }
      compute(&sA[cur][0], &sB[cur][0]);
      if (more) {
        *(v8h*)(&sA[nxt][lofs])     = ra0;
        *(v8h*)(&sA[nxt][lofs + 8]) = ra1;
        *(v8h*)(&sB[nxt][lofs])     = rb0;
        *(v8h*)(&sB[nxt][lofs + 8]) = rb1;
      }
      __syncthreads();
    }
  }

  // ---- epilogue: D layout: VGPR r -> M = r + 8*(lane/16); N = lane&15 ----
  const int mbase = m0 + (wave << 4) + ((lane >> 4) << 3);
  const int ncol  = n0 + (lane & 15);
#pragma unroll
  for (int j = 0; j < 4; ++j) {
    int n = ncol + j * 16;
    if (n >= N) continue;
    int b = n / HW, hw = n % HW;
    float* dst = C + (size_t)b * M * HW + hw;
    v8f acc = (j == 0) ? acc0 : (j == 1) ? acc1 : (j == 2) ? acc2 : acc3;
#pragma unroll
    for (int r = 0; r < 8; ++r) {
      int m = mbase + r;
      if (m < M) {
        float v = acc[r];
        if (bias) v += bias[m];
        dst[(size_t)m * HW] = v;
      }
    }
  }
}

// ---------------------------------------------------------------------------
// Host orchestration
// ---------------------------------------------------------------------------
extern "C" void kernel_launch(void* const* d_in, const int* in_sizes, int n_in,
                              void* d_out, int out_size, void* d_ws, size_t ws_size,
                              hipStream_t stream) {
  (void)in_sizes; (void)n_in; (void)out_size; (void)ws_size;
  const float* inp   = (const float*)d_in[0];
  const float* oc1_w = (const float*)d_in[1];
  const float* oc1_b = (const float*)d_in[2];
  const float* c1_w  = (const float*)d_in[3];
  const float* g1    = (const float*)d_in[4];
  const float* b1    = (const float*)d_in[5];
  const float* oc2_w = (const float*)d_in[6];
  const float* oc2_b = (const float*)d_in[7];
  const float* c2_w  = (const float*)d_in[8];
  const float* g2    = (const float*)d_in[9];
  const float* b2    = (const float*)d_in[10];
  const float* oc3_w = (const float*)d_in[11];
  const float* oc3_b = (const float*)d_in[12];
  const float* c3_w  = (const float*)d_in[13];
  const float* g3    = (const float*)d_in[14];
  const float* b3    = (const float*)d_in[15];
  const float* c4_w  = (const float*)d_in[16];
  const float* g4    = (const float*)d_in[17];
  const float* b4    = (const float*)d_in[18];
  const float* fc1_w = (const float*)d_in[19];
  const float* gfc   = (const float*)d_in[20];
  const float* bfc   = (const float*)d_in[21];
  const float* fc2_w = (const float*)d_in[22];
  float* out = (float*)d_out;

  const int B = 32;
  char* p = (char*)d_ws;
  auto alloc = [&](size_t bytes) -> void* {
    void* r = (void*)p;
    p += (bytes + 255) & ~(size_t)255;
    return r;
  };

  // f16 weights (Kpad = round-up to 32)
  _Float16* oc1h = (_Float16*)alloc((size_t)18 * 32 * 2);
  _Float16* c1h  = (_Float16*)alloc((size_t)64 * 32 * 2);
  _Float16* oc2h = (_Float16*)alloc((size_t)18 * 576 * 2);
  _Float16* c2h  = (_Float16*)alloc((size_t)128 * 576 * 2);
  _Float16* oc3h = (_Float16*)alloc((size_t)18 * 1152 * 2);
  _Float16* c3h  = (_Float16*)alloc((size_t)256 * 1152 * 2);
  _Float16* c4h  = (_Float16*)alloc((size_t)256 * 2304 * 2);
  _Float16* fc1h = (_Float16*)alloc((size_t)1024 * 4096 * 2);
  _Float16* fc2h = (_Float16*)alloc((size_t)10 * 1024 * 2);

  // activations / patches
  float*    sp   = (float*)alloc((size_t)B * 3 * 1024 * 4);
  _Float16* P1   = (_Float16*)alloc((size_t)32768 * 32 * 2);
  float*    off1 = (float*)alloc((size_t)B * 18 * 1024 * 4);
  _Float16* D1   = (_Float16*)alloc((size_t)32768 * 32 * 2);
  float*    y1   = (float*)alloc((size_t)B * 64 * 1024 * 4);
  float*    m1   = (float*)alloc((size_t)B * 64 * 1024 * 4);
  float*    s1   = (float*)alloc((size_t)B * 64 * 1024 * 4);
  float*    o1   = (float*)alloc((size_t)B * 64 * 256 * 4);
  _Float16* P2   = (_Float16*)alloc((size_t)8192 * 576 * 2);
  float*    off2 = (float*)alloc((size_t)B * 18 * 256 * 4);
  _Float16* D2   = (_Float16*)alloc((size_t)8192 * 576 * 2);
  float*    y2   = (float*)alloc((size_t)B * 128 * 256 * 4);
  float*    m2   = (float*)alloc((size_t)B * 128 * 256 * 4);
  float*    s2   = (float*)alloc((size_t)B * 128 * 256 * 4);
  float*    o2   = (float*)alloc((size_t)B * 128 * 64 * 4);
  _Float16* P3   = (_Float16*)alloc((size_t)2048 * 1152 * 2);
  float*    off3 = (float*)alloc((size_t)B * 18 * 64 * 4);
  _Float16* D3   = (_Float16*)alloc((size_t)2048 * 1152 * 2);
  float*    y3   = (float*)alloc((size_t)B * 256 * 64 * 4);
  float*    m3   = (float*)alloc((size_t)B * 256 * 64 * 4);
  float*    s3   = (float*)alloc((size_t)B * 256 * 64 * 4);
  _Float16* P4   = (_Float16*)alloc((size_t)2048 * 2304 * 2);
  float*    y4   = (float*)alloc((size_t)B * 256 * 64 * 4);
  float*    m4   = (float*)alloc((size_t)B * 256 * 64 * 4);
  float*    s4   = (float*)alloc((size_t)B * 256 * 64 * 4);
  _Float16* P5   = (_Float16*)alloc((size_t)32 * 4096 * 2);
  float*    z1   = (float*)alloc((size_t)32 * 1024 * 4);
  float*    mf1  = (float*)alloc((size_t)32 * 1024 * 4);
  _Float16* Sf   = (_Float16*)alloc((size_t)32 * 1024 * 2);
  float*    tmp2 = (float*)alloc((size_t)32 * 10 * 4);
  float*    mean = (float*)alloc((size_t)1024 * 4);
  float*    var  = (float*)alloc((size_t)1024 * 4);

  size_t used = (size_t)(p - (char*)d_ws);

  // zero workspace (membranes + f16 pad lanes) and output
  k_zero<<<2048, 256, 0, stream>>>((float*)d_ws, used / 4);
  k_zero<<<2, 256, 0, stream>>>(out, 320);

  auto cvt = [&](const float* w, _Float16* o, int M, int K, int Kpad) {
    int n = M * Kpad;
    k_cvt_w<<<(n + 255) / 256, 256, 0, stream>>>(w, o, M, K, Kpad);
  };
  cvt(oc1_w, oc1h, 18, 27, 32);
  cvt(c1_w,  c1h,  64, 27, 32);
  cvt(oc2_w, oc2h, 18, 576, 576);
  cvt(c2_w,  c2h,  128, 576, 576);
  cvt(oc3_w, oc3h, 18, 1152, 1152);
  cvt(c3_w,  c3h,  256, 1152, 1152);
  cvt(c4_w,  c4h,  256, 2304, 2304);
  cvt(fc1_w, fc1h, 1024, 4096, 4096);
  cvt(fc2_w, fc2h, 10, 1024, 1024);

  auto gemm = [&](const _Float16* A, const _Float16* Bp, const float* bias,
                  float* C, int M, int N, int Kpad, int HW) {
    dim3 g((N + GT_N - 1) / GT_N, (M + GT_M - 1) / GT_M);
    k_gemm_wmma<<<g, 128, 0, stream>>>(A, Bp, bias, C, M, N, Kpad, HW);
  };

  for (int t = 0; t < 20; ++t) {
    // ---- input spikes ----
    k_spike_in<<<(98304 + 255) / 256, 256, 0, stream>>>(inp, sp, 98304, t);

    // ---- layer 1 (3 -> 64 @ 32x32) ----
    k_im2col<<<(98304 + 255) / 256, 256, 0, stream>>>(sp, P1, B, 3, 32, 32, 32);
    gemm(oc1h, P1, oc1_b, off1, 18, 32768, 32, 1024);
    k_deform<<<(294912 + 255) / 256, 256, 0, stream>>>(sp, off1, D1, B, 3, 32, 32, 32);
    gemm(c1h, D1, nullptr, y1, 64, 32768, 32, 1024);
    k_bnstats<<<64, 256, 0, stream>>>(y1, mean, var, 64, 32 * 1024, 1024);
    {
      int n = B * 64 * 1024;
      k_mem2d<<<(n + 255) / 256, 256, 0, stream>>>(y1, mean, var, g1 + t * 64,
                                                   b1 + t * 64, m1, s1, 64, 1024, n);
      int np = B * 64 * 256;
      k_pool<<<(np + 255) / 256, 256, 0, stream>>>(s1, o1, 32, 32, np);
    }

    // ---- layer 2 (64 -> 128 @ 16x16) ----
    k_im2col<<<(524288 + 255) / 256, 256, 0, stream>>>(o1, P2, B, 64, 16, 16, 576);
    gemm(oc2h, P2, oc2_b, off2, 18, 8192, 576, 256);
    k_deform<<<(73728 + 255) / 256, 256, 0, stream>>>(o1, off2, D2, B, 64, 16, 16, 576);
    gemm(c2h, D2, nullptr, y2, 128, 8192, 576, 256);
    k_bnstats<<<128, 256, 0, stream>>>(y2, mean, var, 128, 32 * 256, 256);
    {
      int n = B * 128 * 256;
      k_mem2d<<<(n + 255) / 256, 256, 0, stream>>>(y2, mean, var, g2 + t * 128,
                                                   b2 + t * 128, m2, s2, 128, 256, n);
      int np = B * 128 * 64;
      k_pool<<<(np + 255) / 256, 256, 0, stream>>>(s2, o2, 16, 16, np);
    }

    // ---- layer 3 (128 -> 256 @ 8x8, no pool) ----
    k_im2col<<<(262144 + 255) / 256, 256, 0, stream>>>(o2, P3, B, 128, 8, 8, 1152);
    gemm(oc3h, P3, oc3_b, off3, 18, 2048, 1152, 64);
    k_deform<<<(18432 + 255) / 256, 256, 0, stream>>>(o2, off3, D3, B, 128, 8, 8, 1152);
    gemm(c3h, D3, nullptr, y3, 256, 2048, 1152, 64);
    k_bnstats<<<256, 256, 0, stream>>>(y3, mean, var, 256, 32 * 64, 64);
    {
      int n = B * 256 * 64;
      k_mem2d<<<(n + 255) / 256, 256, 0, stream>>>(y3, mean, var, g3 + t * 256,
                                                   b3 + t * 256, m3, s3, 256, 64, n);
    }

    // ---- layer 4 (256 -> 256 @ 8x8, then pool -> 4x4) ----
    k_im2col<<<(524288 + 255) / 256, 256, 0, stream>>>(s3, P4, B, 256, 8, 8, 2304);
    gemm(c4h, P4, nullptr, y4, 256, 2048, 2304, 64);
    k_bnstats<<<256, 256, 0, stream>>>(y4, mean, var, 256, 32 * 64, 64);
    {
      int n = B * 256 * 64;
      k_mem2d<<<(n + 255) / 256, 256, 0, stream>>>(y4, mean, var, g4 + t * 256,
                                                   b4 + t * 256, m4, s4, 256, 64, n);
      int np = B * 256 * 16;
      k_pool_fc<<<(np + 255) / 256, 256, 0, stream>>>(s4, P5, 256, 8, 8, 4096, np);
    }

    // ---- FC head ----
    gemm(fc1h, P5, nullptr, z1, 1024, 32, 4096, 1);
    k_bnstats1d<<<(1024 + 255) / 256, 256, 0, stream>>>(z1, mean, var, 1024, 32);
    {
      int n = 32 * 1024;
      k_memfc<<<(n + 255) / 256, 256, 0, stream>>>(z1, mean, var, gfc + t * 1024,
                                                   bfc + t * 1024, mf1, Sf, 1024, n);
    }
    gemm(fc2h, Sf, nullptr, tmp2, 10, 32, 1024, 1);
    k_accout<<<2, 256, 0, stream>>>(tmp2, out, 320);
  }
}
